// STRN_52450140619510
// MI455X (gfx1250) — matrix operand order repaired and checked by reading, not compiled, not run on hardware
//
#include <hip/hip_runtime.h>

// ---------------------------------------------------------------------------
// CDNA5 (gfx1250) wave32 WMMA implementation of the word-cell routing model.
// All GEMMs use v_wmma_f32_16x16x32_bf16 (bf16 in, f32 accumulate).
// ---------------------------------------------------------------------------

typedef __attribute__((ext_vector_type(16))) __bf16 v16bf;
typedef __attribute__((ext_vector_type(8)))  float  v8f;

#define N_WRD 2000
#define EMB   32
#define VDIM  256   // 16x16 view grid
#define BSZ   64
#define CSIZE 128
#define MROWS (BSZ * N_WRD)   // 128000

union FragU {
    v16bf v;
    unsigned u[8];
    unsigned short us[16];
};

__device__ __forceinline__ unsigned short f2bf(float f) {
    unsigned u = __builtin_bit_cast(unsigned, f);
    unsigned r = u + 0x7fffu + ((u >> 16) & 1u);   // round-to-nearest-even
    return (unsigned short)(r >> 16);
}

__device__ __forceinline__ v8f vzero() {
    v8f z;
#pragma unroll
    for (int i = 0; i < 8; ++i) z[i] = 0.0f;
    return z;
}

__device__ __forceinline__ v8f wmma_bf16(v8f c, v16bf a, v16bf b) {
    // (neg_a, A, neg_b, B, c_mod, C, reuse_a, reuse_b)
    return __builtin_amdgcn_wmma_f32_16x16x32_bf16(false, a, false, b,
                                                   (short)0, c, false, false);
}

// A-matrix (16x32 bf16) per-lane K offset of VGPR pair j (contiguous, matches B):
// lanes 0-15 hold K {0..15}, lanes 16-31 hold K {16..31}.
__device__ __forceinline__ int a_kpair(int j, int lh) {
    return 2 * j + 16 * lh;
}

__device__ __forceinline__ float hsum16(float x) {
#pragma unroll
    for (int m = 1; m < 16; m <<= 1) x += __shfl_xor(x, m, 32);
    return x;
}
__device__ __forceinline__ float hmax16(float x) {
#pragma unroll
    for (int m = 1; m < 16; m <<= 1) x = fmaxf(x, __shfl_xor(x, m, 32));
    return x;
}

// ---------------------------------------------------------------------------
// Pack a row-major f32 K x N weight matrix into per-lane WMMA B-fragments:
// dst layout [ntile][kstep][lane][16 bf16]  (32 B per lane => b128 frag loads)
// B-matrix layout: lane l holds column n = ntile*16 + (l&15);
//   elems e=0..15 hold K = kstep*32 + 16*(l>>4) + e.
// ---------------------------------------------------------------------------
__global__ void k_pack_b(const float* __restrict__ src,
                         unsigned short* __restrict__ dst, int K, int N) {
    int gid = blockIdx.x * 256 + threadIdx.x;
    if (gid >= K * N) return;
    int e     = gid & 15;
    int lane  = (gid >> 4) & 31;
    int t     = gid >> 9;
    int kst   = K >> 5;
    int kstep = t % kst;
    int ntile = t / kst;
    int k = kstep * 32 + ((lane >> 4) << 4) + e;
    int n = ntile * 16 + (lane & 15);
    dst[gid] = f2bf(src[k * N + n]);
}

// ---------------------------------------------------------------------------
// View-space embedding: vs[v][e] for the 16x16 pixel grid, emitted directly
// as packed B-fragments for the relation GEMM (K=EMB=32, N=VDIM=256).
// ---------------------------------------------------------------------------
__global__ void k_vs(const float* __restrict__ w1, const float* __restrict__ b1,
                     const float* __restrict__ w2, const float* __restrict__ b2,
                     unsigned short* __restrict__ vsp) {
    int v = threadIdx.x;                      // 256 threads
    float x = -1.0f + 2.0f * (float)(v >> 4) / 15.0f;
    float y = -1.0f + 2.0f * (float)(v & 15) / 15.0f;
    float row[EMB];
#pragma unroll
    for (int e = 0; e < EMB; ++e) row[e] = b2[e];
    for (int t = 0; t < 128; ++t) {
        float h = fmaxf(x * w1[t] + y * w1[128 + t] + b1[t], 0.0f);
#pragma unroll
        for (int e = 0; e < EMB; ++e) row[e] += h * w2[t * EMB + e];
    }
    int ntile = v >> 4, lp = v & 15;
#pragma unroll
    for (int e = 0; e < 16; ++e)
        vsp[(ntile * 32 + lp) * 16 + e] = f2bf(row[e]);
#pragma unroll
    for (int e = 0; e < 16; ++e)
        vsp[(ntile * 32 + 16 + lp) * 16 + e] = f2bf(row[16 + e]);
}

// h0 = relu(v @ w2c_w1 + b1), stored bf16 row-major (64 x 512)
__global__ void k_h0(const float* __restrict__ vin, const float* __restrict__ w1,
                     const float* __restrict__ b1, unsigned short* __restrict__ h0) {
    int b = blockIdx.x, o = threadIdx.x;      // 64 blocks x 512 threads
    float s = b1[o];
#pragma unroll
    for (int k = 0; k < 7; ++k) s += vin[b * 7 + k] * w1[k * 512 + o];
    h0[b * 512 + o] = f2bf(fmaxf(s, 0.0f));
}

// wcode = h0 @ w2c_w2 + b2 : M=64 (grid.y tiles), N=6000 (grid.x), K=512.
__global__ __launch_bounds__(32) void k_wcode(
    const unsigned short* __restrict__ h0, const v16bf* __restrict__ w2p,
    const float* __restrict__ b2, float* __restrict__ wcode) {
    int nt = blockIdx.x, mt = blockIdx.y;
    int lane = threadIdx.x, lm = lane & 15, lh = lane >> 4;
    const unsigned* h0u = (const unsigned*)h0;   // bf16 pairs
    v8f acc = vzero();
    for (int s = 0; s < 16; ++s) {
        FragU a;
#pragma unroll
        for (int j = 0; j < 8; ++j)
            a.u[j] = h0u[(mt * 16 + lm) * 256 + s * 16 + (a_kpair(j, lh) >> 1)];
        acc = wmma_bf16(acc, a.v, w2p[(nt * 16 + s) * 32 + lane]);
    }
    float bias = b2[nt * 16 + lm];
#pragma unroll
    for (int j = 0; j < 8; ++j)
        wcode[(mt * 16 + j + 8 * lh) * 6000 + nt * 16 + lm] = acc[j] + bias;
}

// h1 = relu(wcode @ fc1_w + fc1_b), K=3 -> VALU; bf16 row-major (128000 x 256)
__global__ void k_fc1(const float* __restrict__ wcode, const float* __restrict__ fw,
                      const float* __restrict__ fb, unsigned short* __restrict__ h1) {
    int m = blockIdx.x, n = threadIdx.x;      // one row per block
    int b = m / N_WRD, nw = m % N_WRD;
    const float* wc = wcode + b * 6000 + nw * 3;
    float s = wc[0] * fw[n] + wc[1] * fw[256 + n] + wc[2] * fw[512 + n] + fb[n];
    h1[m * 256 + n] = f2bf(fmaxf(s, 0.0f));
}

// ---------------------------------------------------------------------------
// Fused: fc2(+ReLU) -> {sigmoid(fca), fce} -> relation(vs) -> softmax*act
// Block = 64 threads = 2 waves, each wave owns a 16-row tile.
// route (bf16, row-major 256 cols) is written over h1 (same rows, same size).
// ---------------------------------------------------------------------------
__global__ __launch_bounds__(64) void k_fused(
    unsigned short* __restrict__ h1route,          // bf16 in (h1) / out (route)
    const v16bf* __restrict__ fc2p, const float* __restrict__ fc2_b,
    const float* __restrict__ fca_w, const float* __restrict__ fca_b,
    const v16bf* __restrict__ fcep, const float* __restrict__ fce_b,
    const v16bf* __restrict__ vsp) {
    __shared__ unsigned short lds_h2[2 * 16 * 256];   // per-wave h2 tiles (bf16)
    __shared__ unsigned short lds_emb[2 * 16 * EMB];  // per-wave cs_emb tiles

    const int wv = threadIdx.x >> 5;
    const int lane = threadIdx.x & 31, lm = lane & 15, lh = lane >> 4;
    const int m0 = blockIdx.x * 32 + wv * 16;

    // ---- fc2: (16 x 256) = h1_tile(16x256) @ fc2_w(256x256) --------------
    v8f acc[16];
#pragma unroll
    for (int t = 0; t < 16; ++t) acc[t] = vzero();
    const unsigned* h1u = (const unsigned*)h1route;
    for (int s = 0; s < 8; ++s) {
        FragU a;
#pragma unroll
        for (int j = 0; j < 8; ++j)
            a.u[j] = h1u[(m0 + lm) * 128 + s * 16 + (a_kpair(j, lh) >> 1)];
#pragma unroll
        for (int t = 0; t < 16; ++t)
            acc[t] = wmma_bf16(acc[t], a.v, fc2p[(t * 8 + s) * 32 + lane]);
    }
#pragma unroll
    for (int t = 0; t < 16; ++t) {
        float nb = fc2_b[t * 16 + lm];
#pragma unroll
        for (int j = 0; j < 8; ++j) acc[t][j] = fmaxf(acc[t][j] + nb, 0.0f);
    }

    // ---- activation = sigmoid(h2 @ fca_w + fca_b), per-row reduce --------
    float fcav[16];
#pragma unroll
    for (int t = 0; t < 16; ++t) fcav[t] = fca_w[t * 16 + lm];
    float act[8];
#pragma unroll
    for (int j = 0; j < 8; ++j) {
        float s = 0.0f;
#pragma unroll
        for (int t = 0; t < 16; ++t) s += acc[t][j] * fcav[t];
        s = hsum16(s) + fca_b[0];
        act[j] = 1.0f / (1.0f + __expf(-s));
    }

    // ---- stash h2 (bf16) in LDS so it can be re-read in A-fragment order -
    unsigned short* myh2 = lds_h2 + wv * 16 * 256;
#pragma unroll
    for (int t = 0; t < 16; ++t)
#pragma unroll
        for (int j = 0; j < 8; ++j)
            myh2[(j + 8 * lh) * 256 + t * 16 + lm] = f2bf(acc[t][j]);
    __syncthreads();

    // ---- cs_emb = h2 @ fce_w + fce_b : (16 x 32), K=256 ------------------
    v8f emb[2] = {vzero(), vzero()};
    for (int s = 0; s < 8; ++s) {
        FragU a;
#pragma unroll
        for (int j = 0; j < 8; ++j)
            a.u[j] = *(const unsigned*)&myh2[lm * 256 + s * 32 + a_kpair(j, lh)];
#pragma unroll
        for (int t = 0; t < 2; ++t)
            emb[t] = wmma_bf16(emb[t], a.v, fcep[(t * 8 + s) * 32 + lane]);
    }
    unsigned short* myemb = lds_emb + wv * 16 * EMB;
#pragma unroll
    for (int t = 0; t < 2; ++t) {
        float eb = fce_b[t * 16 + lm];
#pragma unroll
        for (int j = 0; j < 8; ++j)
            myemb[(j + 8 * lh) * EMB + t * 16 + lm] = f2bf(emb[t][j] + eb);
    }
    __syncthreads();

    // ---- relation = cs_emb @ vs^T : (16 x 256), K=32 ---------------------
    v8f rel[16];
#pragma unroll
    for (int t = 0; t < 16; ++t) rel[t] = vzero();
    {
        FragU a;
#pragma unroll
        for (int j = 0; j < 8; ++j)
            a.u[j] = *(const unsigned*)&myemb[lm * EMB + a_kpair(j, lh)];
#pragma unroll
        for (int t = 0; t < 16; ++t)
            rel[t] = wmma_bf16(rel[t], a.v, vsp[t * 32 + lane]);
    }

    // ---- softmax over the 256 view cells, scaled by activation ----------
    float mx[8], sc[8];
#pragma unroll
    for (int j = 0; j < 8; ++j) {
        float m = -3.0e38f;
#pragma unroll
        for (int t = 0; t < 16; ++t) m = fmaxf(m, rel[t][j]);
        mx[j] = hmax16(m);
    }
#pragma unroll
    for (int j = 0; j < 8; ++j) {
        float s = 0.0f;
#pragma unroll
        for (int t = 0; t < 16; ++t) {
            rel[t][j] = __expf(rel[t][j] - mx[j]);
            s += rel[t][j];
        }
        sc[j] = act[j] / hsum16(s);
    }
    // route overwrites h1 (all reads of these rows are done, block-local)
#pragma unroll
    for (int t = 0; t < 16; ++t)
#pragma unroll
        for (int j = 0; j < 8; ++j)
            h1route[(m0 + j + 8 * lh) * 256 + t * 16 + lm] = f2bf(rel[t][j] * sc[j]);
}

// ---------------------------------------------------------------------------
// wrd_cell[b] = view_cell[b] (128 x 256) @ route[b]^T (256 x 2000)
// grid (125 n-tiles, 64 batches), 256 threads = 8 waves (full M=128).
// ---------------------------------------------------------------------------
__global__ __launch_bounds__(256) void k_out(
    const float* __restrict__ view_cell, const unsigned short* __restrict__ route,
    float* __restrict__ out) {
    __shared__ unsigned short lds_r[16 * 256];     // route tile (16 rows x 256 v)
    int nt = blockIdx.x, b = blockIdx.y, tid = threadIdx.x;
    int nbase = nt * 16;

    const unsigned* ru = (const unsigned*)route;
    unsigned* lu = (unsigned*)lds_r;
#pragma unroll
    for (int i = tid; i < 16 * 128; i += 256) {
        int r = i >> 7, c = i & 127;
        lu[i] = ru[(b * N_WRD + nbase + r) * 128 + c];
    }
    __syncthreads();

    int wv = tid >> 5, lane = tid & 31, lm = lane & 15, lh = lane >> 4;
    v8f acc = vzero();
    for (int s = 0; s < 8; ++s) {
        FragU a;
#pragma unroll
        for (int j = 0; j < 8; ++j) {
            int kp = a_kpair(j, lh);
            const float2 p = *(const float2*)(view_cell +
                ((size_t)(b * CSIZE + wv * 16 + lm) * 256 + s * 32 + kp));
            a.us[2 * j]     = f2bf(p.x);
            a.us[2 * j + 1] = f2bf(p.y);
        }
        FragU bb;
#pragma unroll
        for (int j = 0; j < 8; ++j)
            bb.u[j] = lu[lm * 128 + s * 16 + 8 * lh + j];
        acc = wmma_bf16(acc, a.v, bb.v);
    }
#pragma unroll
    for (int j = 0; j < 8; ++j)
        out[((size_t)(b * CSIZE + wv * 16 + j + 8 * lh)) * N_WRD + nbase + lm] = acc[j];
}

// ---------------------------------------------------------------------------
extern "C" void kernel_launch(void* const* d_in, const int* in_sizes, int n_in,
                              void* d_out, int out_size, void* d_ws, size_t ws_size,
                              hipStream_t stream) {
    const float* view_cell = (const float*)d_in[0];
    const float* vin       = (const float*)d_in[1];
    const float* w2c_w1    = (const float*)d_in[2];
    const float* w2c_b1    = (const float*)d_in[3];
    const float* w2c_w2    = (const float*)d_in[4];
    const float* w2c_b2    = (const float*)d_in[5];
    const float* fc1_w     = (const float*)d_in[6];
    const float* fc1_b     = (const float*)d_in[7];
    const float* fc2_w     = (const float*)d_in[8];
    const float* fc2_b     = (const float*)d_in[9];
    const float* fca_w     = (const float*)d_in[10];
    const float* fca_b     = (const float*)d_in[11];
    const float* fce_w     = (const float*)d_in[12];
    const float* fce_b     = (const float*)d_in[13];
    const float* vse_w1    = (const float*)d_in[14];
    const float* vse_b1    = (const float*)d_in[15];
    const float* vse_w2    = (const float*)d_in[16];
    const float* vse_b2    = (const float*)d_in[17];
    float* out = (float*)d_out;

    char* ws = (char*)d_ws;
    size_t o_h1   = 0;                               // h1 / route (bf16), 65,536,000 B
    size_t o_w2p  = o_h1  + (size_t)MROWS * 256 * 2; // packed w2c_w2, 6,144,000 B
    size_t o_fc2p = o_w2p + (size_t)512 * 6000 * 2;  // packed fc2_w, 131,072 B
    size_t o_fcep = o_fc2p + 256 * 256 * 2;          // packed fce_w, 16,384 B
    size_t o_vsp  = o_fcep + 256 * EMB * 2;          // packed vs,    16,384 B
    size_t o_h0   = o_vsp  + EMB * VDIM * 2;         // h0 bf16,      65,536 B
    size_t o_wc   = o_h0   + 64 * 512 * 2;           // wcode f32,    1,536,000 B

    unsigned short* h1    = (unsigned short*)(ws + o_h1);
    unsigned short* w2p   = (unsigned short*)(ws + o_w2p);
    unsigned short* fc2p  = (unsigned short*)(ws + o_fc2p);
    unsigned short* fcep  = (unsigned short*)(ws + o_fcep);
    unsigned short* vsp   = (unsigned short*)(ws + o_vsp);
    unsigned short* h0    = (unsigned short*)(ws + o_h0);
    float*          wcode = (float*)(ws + o_wc);

    k_pack_b<<<(512 * 6000) / 256, 256, 0, stream>>>(w2c_w2, w2p, 512, 6000);
    k_pack_b<<<(256 * 256) / 256, 256, 0, stream>>>(fc2_w, fc2p, 256, 256);
    k_pack_b<<<(256 * EMB) / 256, 256, 0, stream>>>(fce_w, fcep, 256, EMB);
    k_vs<<<1, 256, 0, stream>>>(vse_w1, vse_b1, vse_w2, vse_b2, vsp);
    k_h0<<<64, 512, 0, stream>>>(vin, w2c_w1, w2c_b1, h0);
    k_wcode<<<dim3(375, 4), 32, 0, stream>>>(h0, (const v16bf*)w2p, w2c_b2, wcode);
    k_fc1<<<MROWS, 256, 0, stream>>>(wcode, fc1_w, fc1_b, h1);
    k_fused<<<MROWS / 32, 64, 0, stream>>>(h1, (const v16bf*)fc2p, fc2_b,
                                           fca_w, fca_b, (const v16bf*)fcep,
                                           fce_b, (const v16bf*)vsp);
    k_out<<<dim3(125, 64), 256, 0, stream>>>(view_cell, h1, out);
}
